// CPMKernel_60215441490497
// MI455X (gfx1250) — matrix operand order repaired and checked by reading, not compiled, not run on hardware
//
#include <hip/hip_runtime.h>
#include <hip/hip_bf16.h>

#define H 2048
#define W 2048
#define HW (H * W)
#define NSTEPS 128
#define NTYPES 8
#define BPR 8            // blocks per row in kernel A (W / 256)

typedef __attribute__((ext_vector_type(16))) _Float16 v16h;
typedef __attribute__((ext_vector_type(8)))  float    v8f;

// ---------------------------------------------------------------------------
// Threefry-2x32 (20 rounds) — counter-based RNG mirroring JAX's scheme.
// ---------------------------------------------------------------------------
__device__ __forceinline__ void tf2x32(unsigned k0, unsigned k1,
                                       unsigned c0, unsigned c1,
                                       unsigned& o0, unsigned& o1) {
  unsigned ks[3] = {k0, k1, 0x1BD11BDAu ^ k0 ^ k1};
  unsigned x0 = c0 + k0, x1 = c1 + k1;
  const int R[8] = {13, 15, 26, 6, 17, 29, 16, 24};
  #pragma unroll
  for (int i = 0; i < 5; ++i) {
    #pragma unroll
    for (int r = 0; r < 4; ++r) {
      x0 += x1;
      int rot = R[(i & 1) * 4 + r];
      x1 = (x1 << rot) | (x1 >> (32 - rot));
      x1 ^= x0;
    }
    x0 += ks[(i + 1) % 3];
    x1 += ks[(i + 2) % 3] + (unsigned)(i + 1);
  }
  o0 = x0; o1 = x1;
}

__device__ __forceinline__ float u01(unsigned k0, unsigned k1) {
  unsigned a, b;
  tf2x32(k0, k1, 0u, 1u, a, b);
  return (float)(a >> 8) * (1.0f / 16777216.0f);
}

// ---------------------------------------------------------------------------
// Kernel A: copy state, stage mask, per-block row-sum partials, and the
// adhesion energy as a type-pair histogram via V_WMMA_F32_16X16X32_F16.
//   C[m][n] += Sigma_k onehot(tL[k])[m] * onehot(tR[k])[n]   (exact f32 counts)
//   energy  = Sigma_{m,n} C[m][n] * J[m][n]
// A operand 16x32 f16 layout: lane -> M=lane&15; K = (e<8?0:16)+(lane>=16?8:0)+(e&7)
// B operand 32x16 f16 layout: lane -> N=lane&15; K = (e<8?0:16)+(lane>=16?8:0)+(e&7)
// ---------------------------------------------------------------------------
__global__ void __launch_bounds__(256)
cpm_init_kernel(const int* __restrict__ cpm,
                const float* __restrict__ mask_in,
                const float* __restrict__ J,
                float* __restrict__ out,       // [ids | types | energy | accept]
                float* __restrict__ mask_ws,   // [H*W]
                float* __restrict__ rowpart,   // [H*BPR]
                float* __restrict__ eparts) {  // [H*BPR]
  const int x    = blockIdx.x * 256 + threadIdx.x;
  const int y    = blockIdx.y;
  const int idx  = y * W + x;
  const int lane = threadIdx.x & 31;
  const int wid  = threadIdx.x >> 5;

  const int id = cpm[idx];
  const int tp = cpm[HW + idx];
  out[idx]      = (float)id;
  out[HW + idx] = (float)tp;
  const float m = mask_in[idx];
  mask_ws[idx]  = m;

  // wave partial of the row sum
  float rs = m;
  #pragma unroll
  for (int off = 16; off >= 1; off >>= 1) rs += __shfl_down(rs, off, 32);

  // neighbor pair data (right along row, down along column; periodic)
  const int xr  = (x + 1) & (W - 1);
  const int yd  = (y + 1) & (H - 1);
  __builtin_prefetch(cpm + yd * W + x, 0, 3);           // global_prefetch_b8
  const int idr = cpm[y * W + xr];
  const int tpr = cpm[HW + y * W + xr];
  const int idd = cpm[yd * W + x];
  const int tpd = cpm[HW + yd * W + x];

  const int avalR = (id != idr) ? tp : 255;             // 255 -> zero one-hot row
  const int avalD = (id != idd) ? tp : 255;

  const int mrow  = lane & 15;
  const int abase = (lane >= 16) ? 8 : 0;
  const int bbase = (lane >= 16) ? 16 : 0;

  v8f acc = {};
  v16h a, b;
  #pragma unroll
  for (int e = 0; e < 16; ++e) {
    const int k  = ((e < 8) ? 0 : 16) + abase + (e & 7);
    const int ta = __shfl(avalR, k, 32);
    const int tb = __shfl(tpr, k, 32);
    a[e] = (ta == mrow) ? (_Float16)1.0f : (_Float16)0.0f;
    b[e] = (tb == mrow) ? (_Float16)1.0f : (_Float16)0.0f;
  }
  acc = __builtin_amdgcn_wmma_f32_16x16x32_f16(false, a, false, b,
                                               (short)0, acc, false, false);
  #pragma unroll
  for (int e = 0; e < 16; ++e) {
    const int k  = ((e < 8) ? 0 : 16) + abase + (e & 7);
    const int ta = __shfl(avalD, k, 32);
    const int tb = __shfl(tpd, k, 32);
    a[e] = (ta == mrow) ? (_Float16)1.0f : (_Float16)0.0f;
    b[e] = (tb == mrow) ? (_Float16)1.0f : (_Float16)0.0f;
  }
  acc = __builtin_amdgcn_wmma_f32_16x16x32_f16(false, a, false, b,
                                               (short)0, acc, false, false);

  // C layout: VGPR r, lanes 0-15 -> element (M=r, N=lane). Types < 8, so all
  // nonzero counts sit in lanes 0..7, VGPRs 0..7.
  float part = 0.0f;
  if (lane < NTYPES) {
    #pragma unroll
    for (int r = 0; r < NTYPES; ++r) part += acc[r] * J[r * NTYPES + lane];
  }
  #pragma unroll
  for (int off = 16; off >= 1; off >>= 1) part += __shfl_down(part, off, 32);

  // deterministic per-block partials (no float atomics)
  __shared__ float wE[8], wM[8];
  if (lane == 0) { wE[wid] = part; wM[wid] = rs; }
  __syncthreads();
  if (threadIdx.x == 0) {
    float se = 0.0f, sm = 0.0f;
    #pragma unroll
    for (int i = 0; i < 8; ++i) { se += wE[i]; sm += wM[i]; }
    const int bid = y * BPR + blockIdx.x;
    eparts[bid]  = se;
    rowpart[bid] = sm;
  }
}

// ---------------------------------------------------------------------------
// Kernel B: the sequential 128-step Metropolis chain in one workgroup.
// ---------------------------------------------------------------------------
__device__ __forceinline__ float block_scan(float v, float* sbuf, int tid) {
  __syncthreads();
  sbuf[tid] = v;
  __syncthreads();
  #pragma unroll
  for (int off = 1; off < 256; off <<= 1) {
    float t = (tid >= off) ? sbuf[tid - off] : 0.0f;
    __syncthreads();
    sbuf[tid] += t;
    __syncthreads();
  }
  return sbuf[tid];
}

__device__ __forceinline__ float local_energy(const float* ids, const float* tys,
                                              const float* J, int x, int y,
                                              float sid, int stp) {
  const int DX[4] = {-1, 1, 0, 0};
  const int DY[4] = {0, 0, -1, 1};
  float e = 0.0f;
  #pragma unroll
  for (int i = 0; i < 4; ++i) {
    const int xs = (x + DX[i] + H) % H;
    const int ys = (y + DY[i] + W) % W;
    const float nid = ids[xs * W + ys];
    const int   ntp = (int)tys[xs * W + ys];
    if (nid != sid) e += J[stp * NTYPES + ntp];
  }
  return e;
}

__global__ void __launch_bounds__(256)
cpm_steps_kernel(const float* __restrict__ temps,
                 const float* __restrict__ J,
                 float* __restrict__ out,        // [ids | types | energy | accept]
                 float* __restrict__ mask_ws,
                 const float* __restrict__ rowpart,
                 const float* __restrict__ eparts) {
  __shared__ float lrow[H];        // live per-row boundary sums
  __shared__ float sbuf[256];
  __shared__ float s_e, s_res;
  __shared__ int   s_row, s_col, s_nx, s_ny;

  const int tid = threadIdx.x;
  float* ids = out;
  float* tys = out + HW;
  float* energy_out = out + 2 * HW;
  float* accept_out = out + 2 * HW + NSTEPS;

  // deterministic row sums: fixed-order reduction of the 8 per-row partials
  for (int r = tid; r < H; r += 256) {
    float s = 0.0f;
    #pragma unroll
    for (int b = 0; b < BPR; ++b) s += rowpart[r * BPR + b];
    lrow[r] = s;
  }

  // deterministic total energy: each thread sums 64 partials in order, scan.
  float es = 0.0f;
  for (int i = 0; i < (H * BPR) / 256; ++i) es += eparts[tid * ((H * BPR) / 256) + i];
  (void)block_scan(es, sbuf, tid);
  if (tid == 0) s_e = sbuf[255] / temps[0];   // e0 = total_energy / T[0]

  // root key = jax.random.key(42); keys = split(root, NSTEPS)
  const unsigned RK0 = 0u, RK1 = 42u;

  const int NX8[8] = {-1, -1, -1, 0, 0, 1, 1, 1};
  const int NY8[8] = {-1, 0, 1, -1, 1, -1, 0, 1};
  const int CX5[5] = {0, -1, 1, 0, 0};
  const int CY5[5] = {0, 0, 0, -1, 1};

  for (int s = 0; s < NSTEPS; ++s) {
    __syncthreads();     // lrow / mask updates from previous step visible

    // per-step key, split into (site, neighbor, accept) keys
    unsigned sk0, sk1; tf2x32(RK0, RK1, (unsigned)s, (unsigned)(NSTEPS + s), sk0, sk1);
    unsigned a0, a1, b0, b1, c0, c1;
    tf2x32(sk0, sk1, 0u, 3u, a0, a1);
    tf2x32(sk0, sk1, 1u, 4u, b0, b1);
    tf2x32(sk0, sk1, 2u, 5u, c0, c1);
    const float su = u01(a0, a1);
    unsigned jb0, jb1; tf2x32(b0, b1, 0u, 1u, jb0, jb1);
    const int  j  = (int)(jb0 & 7u);
    const float ua = u01(c0, c1);

    if (tid == 0) { s_row = H - 1; s_res = 0.0f; }

    // ---- pick row proportional to row sums (inverse CDF) ----
    float lv[8], tsum = 0.0f;
    #pragma unroll
    for (int i = 0; i < 8; ++i) { lv[i] = lrow[tid * 8 + i]; tsum += lv[i]; }
    float inc = block_scan(tsum, sbuf, tid);
    const float total = sbuf[255];
    const float u = su * total;
    float exc = inc - tsum;
    if (u >= exc && u < inc) {
      float cum = exc;
      #pragma unroll
      for (int i = 0; i < 8; ++i) {
        if (u < cum + lv[i]) { s_row = tid * 8 + i; s_res = u - cum; break; }
        cum += lv[i];
      }
    }
    __syncthreads();
    const int row = s_row;
    const float resid = s_res;
    if (tid == 0) s_col = W - 1;

    // ---- pick column within the row ----
    float cv[8]; float csum = 0.0f;
    __builtin_prefetch(&mask_ws[row * W + tid * 8], 0, 1);
    #pragma unroll
    for (int i = 0; i < 8; ++i) { cv[i] = mask_ws[row * W + tid * 8 + i]; csum += cv[i]; }
    inc = block_scan(csum, sbuf, tid);
    exc = inc - csum;
    if (resid >= exc && resid < inc) {
      float cum = exc;
      #pragma unroll
      for (int i = 0; i < 8; ++i) {
        if (resid < cum + cv[i]) { s_col = tid * 8 + i; break; }
        cum += cv[i];
      }
    }
    __syncthreads();

    // ---- Metropolis update (single thread) ----
    if (tid == 0) {
      const int sx = s_row, sy = s_col;
      const int nx = (sx + NX8[j] + H) % H;
      const int ny = (sy + NY8[j] + W) % W;
      const float sid = ids[sx * W + sy];
      const int   stp = (int)tys[sx * W + sy];
      const float oid = ids[nx * W + ny];
      const int   otp = (int)tys[nx * W + ny];
      const float eo = local_energy(ids, tys, J, nx, ny, oid, otp);
      const float en = local_energy(ids, tys, J, nx, ny, sid, stp);
      const float delta = (en - eo) / temps[s];
      const int accept = (delta >= 0.0f) || (ua < __expf(-delta));
      if (accept) {
        ids[nx * W + ny] = sid;
        tys[nx * W + ny] = (float)stp;
        s_e += delta;
      }
      energy_out[s] = s_e;
      accept_out[s] = accept ? 1.0f : 0.0f;
      s_nx = nx; s_ny = ny;
      __threadfence();
    }
    __syncthreads();

    // ---- patch boundary mask + row sums at the 5 cross sites ----
    if (tid < 5) {
      const int xs = (s_nx + CX5[tid] + H) % H;
      const int ys = (s_ny + CY5[tid] + W) % W;
      const float cid = ids[xs * W + ys];
      const int DX[4] = {-1, 1, 0, 0};
      const int DY[4] = {0, 0, -1, 1};
      bool bdr = false;
      #pragma unroll
      for (int i = 0; i < 4; ++i) {
        const int xn = (xs + DX[i] + H) % H;
        const int yn = (ys + DY[i] + W) % W;
        bdr = bdr || (ids[xn * W + yn] != cid);
      }
      const float nb = bdr ? 1.0f : 0.0f;
      const float ob = mask_ws[xs * W + ys];
      if (nb != ob) {
        mask_ws[xs * W + ys] = nb;
        atomicAdd(&lrow[xs], nb - ob);      // ds_add_f32, LDS-local
      }
      __threadfence();
    }
  }
}

// ---------------------------------------------------------------------------
extern "C" void kernel_launch(void* const* d_in, const int* in_sizes, int n_in,
                              void* d_out, int out_size, void* d_ws, size_t ws_size,
                              hipStream_t stream) {
  const int*   cpm     = (const int*)d_in[0];     // [2,H,W] int32
  const float* mask_in = (const float*)d_in[1];   // [H,W]  float32
  const float* temps   = (const float*)d_in[2];   // [128]
  const float* J       = (const float*)d_in[3];   // [8,8]

  float* out     = (float*)d_out;                 // ids | types | energy | accept
  float* mask_ws = (float*)d_ws;                  // H*W
  float* rowpart = mask_ws + HW;                  // H*BPR
  float* eparts  = rowpart + H * BPR;             // H*BPR

  const size_t need = (size_t)(HW + 2 * H * BPR) * sizeof(float);
  if (ws_size < need) return;

  cpm_init_kernel<<<dim3(W / 256, H), 256, 0, stream>>>(
      cpm, mask_in, J, out, mask_ws, rowpart, eparts);
  cpm_steps_kernel<<<1, 256, 0, stream>>>(
      temps, J, out, mask_ws, rowpart, eparts);
}
